// MultiHeadSelfAttention_61143154426488
// MI455X (gfx1250) — matrix operand order repaired and checked by reading, not compiled, not run on hardware
//
#include <hip/hip_runtime.h>
#include <hip/hip_bf16.h>
#include <math.h>

typedef __attribute__((ext_vector_type(16))) _Float16 v16h;
typedef __attribute__((ext_vector_type(8)))  float    v8f;

#define EMBED 1024
#define HEADS 16
#define HDIM  64
#define BATCH 4
#define SEQ   2048
#define MROWS (BATCH * SEQ)   // 8192

static __device__ __forceinline__ v8f wmma_f16(v16h a, v16h b, v8f c) {
  // D = A(16x32 f16) * B(32x16 f16) + C(16x16 f32)
  return __builtin_amdgcn_wmma_f32_16x16x32_f16(false, a, false, b, (short)0, c,
                                                false, false);
}

// A-fragment (16x32 f16, MxK) from a row-major f16 matrix.
// Contiguous k order used here (not AMD's table order): lanes 0-15 -> M=lane, halves 0..15 = K kb..kb+15,
// kb = 16*(lane>=16).
static __device__ __forceinline__ v16h load_a_f16(const _Float16* __restrict__ X,
                                                  int ld, int mbase, int koff) {
  int lane = threadIdx.x & 31;
  const _Float16* p = X + (size_t)(mbase + (lane & 15)) * ld + koff + ((lane >> 4) << 4);
  v16h a;
#pragma unroll
  for (int j = 0; j < 8; ++j) {
    a[j]     = p[j];
    a[j + 8] = p[8 + j];
  }
  return a;
}

// B-fragment (32x16 f16, KxN) from a matrix stored [N][Kld] (column-of-B = row of storage).
// Same contiguous k order as load_a_f16: lane -> column N, halves j -> K = koff + 16*(lane>=16) + j  (contiguous 32B).
static __device__ __forceinline__ v16h load_b(const _Float16* __restrict__ Wt,
                                              int ldk, int nbase, int koff) {
  int lane = threadIdx.x & 31;
  const _Float16* p = Wt + (size_t)(nbase + (lane & 15)) * ldk + koff + ((lane >> 4) << 4);
  return *(const v16h*)p;
}

// ---- Kernel 0: X f32 [M][E] -> f16 [M][E] (one-time convert) ----
__global__ void k_xh(const float* __restrict__ X, _Float16* __restrict__ Xh) {
  size_t e = (size_t)blockIdx.x * blockDim.x + threadIdx.x;  // 0 .. 8M-1
  Xh[e] = (_Float16)X[e];
}

// ---- Kernel 1: f32 [K][N] weights -> f16 [N][K] (transposed) for B-fragment loads ----
__global__ void k_wt(const float* __restrict__ Wq, const float* __restrict__ Wk,
                     const float* __restrict__ Wv, const float* __restrict__ Wo,
                     _Float16* __restrict__ WT) {
  int z = blockIdx.y;  // 0..3 : q,k,v,o
  const float* W = (z == 0) ? Wq : (z == 1) ? Wk : (z == 2) ? Wv : Wo;
  _Float16* dst = WT + (size_t)z * EMBED * EMBED;
  int e = blockIdx.x * blockDim.x + threadIdx.x;   // 0 .. 1M-1
  int n = e >> 10, k = e & 1023;
  dst[(size_t)n * EMBED + k] = (_Float16)W[(size_t)k * EMBED + n];
}

// ---- Kernel 2: QKV projection GEMM. One wave computes a 32x64 tile. ----
// mode 0: Q row-major [B,H,S,64]; mode 1: K row-major; mode 2: V transposed [B,H,64,S].
__global__ void __launch_bounds__(32)
k_qkv(const _Float16* __restrict__ Xh, const _Float16* __restrict__ WT,
      const float* __restrict__ bq, const float* __restrict__ bk,
      const float* __restrict__ bv,
      _Float16* __restrict__ Q, _Float16* __restrict__ Km, _Float16* __restrict__ VT) {
  int mode = blockIdx.z;
  const _Float16* Wt = WT + (size_t)mode * EMBED * EMBED;
  const float* bias = (mode == 0) ? bq : (mode == 1) ? bk : bv;
  int mbase = blockIdx.x * 32;
  int nb = blockIdx.y * 64;

  v8f acc[2][4] = {};
  for (int kt = 0; kt < EMBED; kt += 32) {
    v16h a0 = load_a_f16(Xh, EMBED, mbase, kt);
    v16h a1 = load_a_f16(Xh, EMBED, mbase + 16, kt);
#pragma unroll
    for (int t = 0; t < 4; ++t) {
      v16h b = load_b(Wt, EMBED, nb + t * 16, kt);
      acc[0][t] = wmma_f16(a0, b, acc[0][t]);
      acc[1][t] = wmma_f16(a1, b, acc[1][t]);
    }
  }

  int lane = threadIdx.x & 31;
  int col = lane & 15;
  int rowoff = (lane >> 4) << 3;
#pragma unroll
  for (int mt = 0; mt < 2; ++mt) {
#pragma unroll
    for (int t = 0; t < 4; ++t) {
      int n = nb + t * 16 + col;
      int h = n >> 6, d = n & 63;
      float bb = bias[n];
#pragma unroll
      for (int r = 0; r < 8; ++r) {
        int m = mbase + mt * 16 + rowoff + r;
        int bidx = m >> 11, s = m & 2047;
        _Float16 val = (_Float16)(acc[mt][t][r] + bb);
        if (mode == 0)
          Q[(((size_t)bidx * HEADS + h) * SEQ + s) * HDIM + d] = val;
        else if (mode == 1)
          Km[(((size_t)bidx * HEADS + h) * SEQ + s) * HDIM + d] = val;
        else
          VT[(((size_t)bidx * HEADS + h) * HDIM + d) * SEQ + s] = val;
      }
    }
  }
}

// ---- Kernel 3: fused attention. One wave per (b, h, 16-query tile). ----
__global__ void __launch_bounds__(128)
k_attn(const _Float16* __restrict__ Q, const _Float16* __restrict__ Km,
       const _Float16* __restrict__ VT, _Float16* __restrict__ Cc) {
  __shared__ _Float16 P[4][16][32];   // per-wave P tile (exp scores), f16
  int w = threadIdx.x >> 5;
  int lane = threadIdx.x & 31;
  int wid = blockIdx.x * 4 + w;
  int b = wid >> 11;        // / (HEADS * SEQ/16)
  int rem = wid & 2047;
  int h = rem >> 7;
  int qt = rem & 127;

  const _Float16* Qh = Q + (((size_t)b * HEADS + h) * SEQ + qt * 16) * HDIM;
  const _Float16* Kh = Km + (((size_t)b * HEADS + h) * SEQ) * HDIM;
  const _Float16* Vh = VT + (((size_t)b * HEADS + h) * HDIM) * SEQ;

  // Q A-fragments for d = 0..31 and 32..63, reused across the whole key loop.
  v16h qa0 = load_a_f16(Qh, HDIM, 0, 0);
  v16h qa1 = load_a_f16(Qh, HDIM, 0, 32);

  v8f o[4] = {};
  float rs[8] = {};
  const float scale = 0.125f;  // 1/sqrt(64)
  int rowoff = (lane >> 4) << 3;
  int col = lane & 15;

  for (int kt = 0; kt < SEQ; kt += 32) {
    // scores for two 16-key subtiles -> exp -> LDS (C-layout write)
#pragma unroll
    for (int sub = 0; sub < 2; ++sub) {
      int keyb = kt + sub * 16;
      v16h kb0 = load_b(Kh, HDIM, keyb, 0);
      v16h kb1 = load_b(Kh, HDIM, keyb, 32);
      v8f c = {};
      c = wmma_f16(qa0, kb0, c);
      c = wmma_f16(qa1, kb1, c);
#pragma unroll
      for (int r = 0; r < 8; ++r) {
        float e = __expf(c[r] * scale);
        rs[r] += e;
        P[w][rowoff + r][sub * 16 + col] = (_Float16)e;
      }
    }
    asm volatile("s_wait_dscnt 0" ::: "memory");

    // re-read P as an A-fragment (16 rows x 32 keys)
    v16h pa;
    {
      int kb = rowoff;   // == 8*(lane>=16)
#pragma unroll
      for (int j = 0; j < 8; ++j) {
        pa[j]     = P[w][col][2 * kb + j];
        pa[j + 8] = P[w][col][2 * kb + 8 + j];
      }
    }

    // O += P * V  (V^T rows give contiguous B-fragments)
#pragma unroll
    for (int t = 0; t < 4; ++t) {
      v16h vb = load_b(Vh, SEQ, t * 16, kt);
      o[t] = wmma_f16(pa, vb, o[t]);
    }
  }

  // full row sums: reduce across the 16 lanes holding the same rows
#pragma unroll
  for (int r = 0; r < 8; ++r) {
    float v = rs[r];
    v += __shfl_xor(v, 1);
    v += __shfl_xor(v, 2);
    v += __shfl_xor(v, 4);
    v += __shfl_xor(v, 8);
    rs[r] = v;
  }

  // write normalized O into concat buffer [M][E] at column h*64+d
#pragma unroll
  for (int t = 0; t < 4; ++t) {
    int n = h * 64 + t * 16 + col;
#pragma unroll
    for (int r = 0; r < 8; ++r) {
      size_t m = (size_t)b * SEQ + qt * 16 + rowoff + r;
      Cc[m * EMBED + n] = (_Float16)(o[t][r] / rs[r]);
    }
  }
}

// ---- Kernel 4: output projection GEMM, 32x64 tile per wave, f32 result ----
__global__ void __launch_bounds__(32)
k_out(const _Float16* __restrict__ Cc, const _Float16* __restrict__ WoT,
      const float* __restrict__ bo, float* __restrict__ out) {
  int mbase = blockIdx.x * 32;
  int nb = blockIdx.y * 64;
  v8f acc[2][4] = {};
  for (int kt = 0; kt < EMBED; kt += 32) {
    v16h a0 = load_a_f16(Cc, EMBED, mbase, kt);
    v16h a1 = load_a_f16(Cc, EMBED, mbase + 16, kt);
#pragma unroll
    for (int t = 0; t < 4; ++t) {
      v16h b = load_b(WoT, EMBED, nb + t * 16, kt);
      acc[0][t] = wmma_f16(a0, b, acc[0][t]);
      acc[1][t] = wmma_f16(a1, b, acc[1][t]);
    }
  }
  int lane = threadIdx.x & 31;
  int col = lane & 15;
  int rowoff = (lane >> 4) << 3;
#pragma unroll
  for (int mt = 0; mt < 2; ++mt) {
#pragma unroll
    for (int t = 0; t < 4; ++t) {
      int n = nb + t * 16 + col;
      float bb = bo[n];
#pragma unroll
      for (int r = 0; r < 8; ++r) {
        out[(size_t)(mbase + mt * 16 + rowoff + r) * EMBED + n] = acc[mt][t][r] + bb;
      }
    }
  }
}

extern "C" void kernel_launch(void* const* d_in, const int* in_sizes, int n_in,
                              void* d_out, int out_size, void* d_ws, size_t ws_size,
                              hipStream_t stream) {
  const float* X  = (const float*)d_in[0];
  const float* Wq = (const float*)d_in[1];
  const float* bq = (const float*)d_in[2];
  const float* Wk = (const float*)d_in[3];
  const float* bk = (const float*)d_in[4];
  const float* Wv = (const float*)d_in[5];
  const float* bv = (const float*)d_in[6];
  const float* Wo = (const float*)d_in[7];
  const float* bo = (const float*)d_in[8];
  float* out = (float*)d_out;

  char* ws = (char*)d_ws;
  _Float16* WT  = (_Float16*)ws;                          //  8 MB: 4x [1024][1024] f16
  _Float16* Xh  = (_Float16*)(ws + ((size_t)8  << 20));   // 16 MB: [M][E] f16
  _Float16* Qb  = (_Float16*)(ws + ((size_t)24 << 20));   // 16 MB: [B,H,S,64] f16
  _Float16* Kb  = (_Float16*)(ws + ((size_t)40 << 20));   // 16 MB
  _Float16* VTb = (_Float16*)(ws + ((size_t)56 << 20));   // 16 MB: [B,H,64,S] f16
  _Float16* Ccb = (_Float16*)(ws + ((size_t)72 << 20));   // 16 MB: [M][E] f16

  k_xh  <<<(MROWS * EMBED) / 256, 256, 0, stream>>>(X, Xh);
  k_wt  <<<dim3((EMBED * EMBED) / 256, 4), 256, 0, stream>>>(Wq, Wk, Wv, Wo, WT);
  k_qkv <<<dim3(MROWS / 32, EMBED / 64, 3), 32, 0, stream>>>(Xh, WT, bq, bk, bv,
                                                             Qb, Kb, VTb);
  k_attn<<<(BATCH * HEADS * (SEQ / 16)) / 4, 128, 0, stream>>>(Qb, Kb, VTb, Ccb);
  k_out <<<dim3(MROWS / 32, EMBED / 64), 32, 0, stream>>>(
      Ccb, WT + (size_t)3 * EMBED * EMBED, bo, out);
}
